// CrossSpeciesGNN_46626164965915
// MI455X (gfx1250) — matrix operand order repaired and checked by reading, not compiled, not run on hardware
//
#include <hip/hip_runtime.h>
#include <hip/hip_bf16.h>
#include <stdint.h>

#define HDIM   256
#define HDIM2  512
#define NLAYERS 4
#define FINF   8
#define GFDIM  85
#define NGRAPH 128

typedef __attribute__((ext_vector_type(16))) __bf16 v16bf;
typedef __attribute__((ext_vector_type(8)))  __bf16 v8bf;
typedef __attribute__((ext_vector_type(8)))  float  v8f;

// f32 -> bf16 bits, round-to-nearest-even
__device__ inline unsigned short bfbits(float f) {
    unsigned u = __float_as_uint(f);
    return (unsigned short)((u + 0x7FFFu + ((u >> 16) & 1u)) >> 16);
}
__device__ inline __bf16 bfc(float f) {
    unsigned short s = bfbits(f);
    __bf16 b;
    __builtin_memcpy(&b, &s, 2);
    return b;
}

// ---- gfx1250 async global->LDS copy (bypasses VGPRs, tracked by ASYNCcnt) ----
__device__ inline void async_copy16(unsigned lds_addr, const void* gaddr) {
    asm volatile("global_load_async_to_lds_b128 %0, %1, off"
                 :: "v"(lds_addr), "v"(gaddr) : "memory");
}
__device__ inline void wait_async0() { asm volatile("s_wait_asynccnt 0" ::: "memory"); }
__device__ inline void wait_async4() { asm volatile("s_wait_asynccnt 4" ::: "memory"); }

// block-wide (256 threads) sum reduction of two values
__device__ inline float2 block_reduce2(float v1, float v2) {
    __shared__ float s1[256];
    __shared__ float s2[256];
    const int t = threadIdx.x;
    s1[t] = v1; s2[t] = v2;
    __syncthreads();
    #pragma unroll
    for (int off = 128; off > 0; off >>= 1) {
        if (t < off) { s1[t] += s1[t + off]; s2[t] += s2[t + off]; }
        __syncthreads();
    }
    float2 r; r.x = s1[0]; r.y = s2[0];
    __syncthreads();
    return r;
}

// ---------------------------------------------------------------------------
// Weight prep: W1[L][256][512], W2[L][512][256] -> bf16, transposed to [n][k].
// ---------------------------------------------------------------------------
__global__ void prep_w_kernel(const float* __restrict__ W1, const float* __restrict__ W2,
                              void* W1tv, void* W2tv) {
    __bf16* W1t = (__bf16*)W1tv;
    __bf16* W2t = (__bf16*)W2tv;
    int idx = blockIdx.x * 256 + threadIdx.x;
    if (idx >= NLAYERS * HDIM * HDIM2) return;
    int l = idx >> 17;
    int r = idx & 131071;
    int k1 = r >> 9, n1 = r & 511;   // W1 is [256][512]
    W1t[((size_t)l << 17) + (size_t)n1 * HDIM + k1] = bfc(W1[idx]);
    int k2 = r >> 8, n2 = r & 255;   // W2 is [512][256]
    W2t[((size_t)l << 17) + (size_t)n2 * HDIM2 + k2] = bfc(W2[idx]);
}

// fp32 -> bf16 bulk convert (vectorized: 4 elems / thread)
__global__ void to_bf16_kernel(const float* __restrict__ src, void* dstv, int count4) {
    int i = blockIdx.x * 256 + threadIdx.x;
    if (i >= count4) return;
    const float4 v = ((const float4*)src)[i];
    uint2 p;
    p.x = (unsigned)bfbits(v.x) | ((unsigned)bfbits(v.y) << 16);
    p.y = (unsigned)bfbits(v.z) | ((unsigned)bfbits(v.w) << 16);
    ((uint2*)dstv)[i] = p;
}

// ---------------------------------------------------------------------------
// Input projection: h = relu(LN(x @ W_in + b_in)); one block per node.
// ---------------------------------------------------------------------------
__global__ __launch_bounds__(256) void input_proj_kernel(
    const float* __restrict__ x, const float* __restrict__ W, const float* __restrict__ b,
    const float* __restrict__ g, const float* __restrict__ bb, float* __restrict__ h)
{
    const int n = blockIdx.x, c = threadIdx.x;
    __shared__ float xr[FINF];
    if (c < FINF) xr[c] = x[n * FINF + c];
    __syncthreads();
    float a = b[c];
    #pragma unroll
    for (int f = 0; f < FINF; ++f) a += xr[f] * W[f * HDIM + c];
    float2 s = block_reduce2(a, a * a);
    float m = s.x * (1.f / HDIM);
    float v = s.y * (1.f / HDIM) - m * m;
    float y = g[c] * (a - m) * rsqrtf(v + 1e-5f) + bb[c];
    h[(size_t)n * HDIM + c] = fmaxf(y, 0.f);
}

// agg = (2 + eps[layer]) * h   (self-loop fold)
__global__ void agg_init_kernel(const float* __restrict__ h, const float* __restrict__ eps,
                                int layer, float* __restrict__ agg) {
    size_t i = (size_t)blockIdx.x * HDIM + threadIdx.x;
    agg[i] = (2.0f + eps[layer]) * h[i];
}

// agg[dst] += h[src] * w  — one block per edge; rows stay L2-resident
__global__ void edge_scatter_kernel(const int* __restrict__ ei, const float* __restrict__ ew,
                                    const float* __restrict__ h, float* __restrict__ agg, int E) {
    const int e = blockIdx.x, c = threadIdx.x;
    const int src = ei[e], dst = ei[E + e];
    const float w = ew[e];
    atomicAdd(&agg[(size_t)dst * HDIM + c], h[(size_t)src * HDIM + c] * w);
}

// h = relu(LN(Z)) + h
__global__ __launch_bounds__(256) void ln_res_kernel(const float* __restrict__ Z,
    const float* __restrict__ g, const float* __restrict__ bb, float* __restrict__ h)
{
    const int n = blockIdx.x, c = threadIdx.x;
    const float z = Z[(size_t)n * HDIM + c];
    float2 s = block_reduce2(z, z * z);
    float m = s.x * (1.f / HDIM);
    float v = s.y * (1.f / HDIM) - m * m;
    float y = g[c] * (z - m) * rsqrtf(v + 1e-5f) + bb[c];
    size_t i = (size_t)n * HDIM + c;
    h[i] = fmaxf(y, 0.f) + h[i];
}

// ---------------------------------------------------------------------------
// WMMA GEMM with async double-buffered LDS staging (software pipelined).
// C[M,NN] = act(A[M,KK] @ Bt^T + bias), A and Bt bf16, Bt is [NN][KK].
// Block tile 128x128, 8 waves each 64x32 via 4x2 v_wmma_f32_16x16x32_bf16.
// Steady state: s_wait_asynccnt 4 (tile k ready, tile k+1 in flight), barrier,
// 12 ds_load_b128 into distinct regs, 8 back-to-back WMMAs, barrier, refill.
// Last two K-steps are peeled tails (no refill; final step waits asynccnt 0).
// ---------------------------------------------------------------------------
template<int KK, int NN, bool RELU, bool OUT_BF16>
__global__ __launch_bounds__(256) void gemm_wmma_kernel(
    const void* __restrict__ Av, const void* __restrict__ Btv,
    const float* __restrict__ bias, void* __restrict__ Cv, int M)
{
    constexpr int LDT  = 40;          // 32-wide K tile + 8 bf16 pad (80B rows, 16B aligned)
    constexpr int TILE = 128 * LDT;   // elems per (A or B) tile
    __shared__ __align__(16) __bf16 smem[2][2][TILE];   // [buf][A=0,B=1]
    const __bf16* A  = (const __bf16*)Av;
    const __bf16* Bt = (const __bf16*)Btv;

    const int m0 = blockIdx.x * 128;
    const int n0 = blockIdx.y * 128;
    const int tid   = (int)threadIdx.x;
    const int lane  = tid & 31;
    const int wid   = tid >> 5;
    const int wm    = wid & 1;
    const int wn    = wid >> 1;
    const int lhalf = lane >> 4;
    const int l16   = lane & 15;

    const unsigned smem_base = (unsigned)(uintptr_t)(const void*)&smem[0][0][0];

    v8f zero = {};
    v8f acc[4][2];
    #pragma unroll
    for (int i = 0; i < 4; ++i)
        #pragma unroll
        for (int j = 0; j < 2; ++j) acc[i][j] = zero;

    constexpr int T = KK / 32;        // 8 or 16 K-steps (always >= 4)

    auto issue = [&](int kt, int buf) {
        const int k0 = kt * 32;
        const unsigned ldsA = smem_base + (unsigned)(buf * 2 + 0) * (TILE * 2);
        const unsigned ldsB = smem_base + (unsigned)(buf * 2 + 1) * (TILE * 2);
        #pragma unroll
        for (int i = 0; i < 2; ++i) {                 // A: 512 x 16B chunks
            int c = tid + i * 256;
            int row = c >> 2, sub = c & 3;
            int gr = m0 + row; if (gr >= M) gr = M - 1;
            async_copy16(ldsA + (unsigned)(row * LDT + sub * 8) * 2,
                         (const void*)(A + (size_t)gr * KK + k0 + sub * 8));
        }
        #pragma unroll
        for (int i = 0; i < 2; ++i) {                 // B: 512 x 16B chunks
            int c = tid + i * 256;
            int row = c >> 2, sub = c & 3;
            async_copy16(ldsB + (unsigned)(row * LDT + sub * 8) * 2,
                         (const void*)(Bt + (size_t)(n0 + row) * KK + k0 + sub * 8));
        }
    };

    // all fragment loads into distinct regs, then 8 dependency-free WMMAs
    auto compute = [&](int buf) {
        const __bf16* As = &smem[buf][0][0];
        const __bf16* Bs = &smem[buf][1][0];
        v16bf bfrag[2], afrag[4];
        #pragma unroll
        for (int nf = 0; nf < 2; ++nf) {
            const __bf16* p = &Bs[(wn * 32 + nf * 16 + l16) * LDT + lhalf * 8];
            v8bf lo = *(const v8bf*)p;
            v8bf hi = *(const v8bf*)(p + 16);
            bfrag[nf] = __builtin_shufflevector(lo, hi, 0,1,2,3,4,5,6,7,8,9,10,11,12,13,14,15);
        }
        #pragma unroll
        for (int mf = 0; mf < 4; ++mf) {
            const __bf16* p = &As[(wm * 64 + mf * 16 + l16) * LDT + lhalf * 8];
            v8bf lo = *(const v8bf*)p;          // K {0..7} | {8..15}
            v8bf hi = *(const v8bf*)(p + 16);   // K {16..23} | {24..31}
            afrag[mf] = __builtin_shufflevector(lo, hi, 0,1,2,3,4,5,6,7,8,9,10,11,12,13,14,15);
        }
        #pragma unroll
        for (int mf = 0; mf < 4; ++mf)
            #pragma unroll
            for (int nf = 0; nf < 2; ++nf)
                acc[mf][nf] = __builtin_amdgcn_wmma_f32_16x16x32_bf16(
                    false, afrag[mf], false, bfrag[nf], (short)0, acc[mf][nf], false, false);
    };

    issue(0, 0);
    issue(1, 1);

    // steady state: kt = 0 .. T-3
    #pragma unroll 2
    for (int kt = 0; kt < T - 2; ++kt) {
        wait_async4();                 // tile kt complete (kt+1 in flight)
        __syncthreads();
        compute(kt & 1);
        __syncthreads();
        issue(kt + 2, kt & 1);         // refill freed buffer
    }
    // tail: kt = T-2 (tile in flight behind it), then kt = T-1
    wait_async4();
    __syncthreads();
    compute(T & 1);                    // (T-2) & 1 == T & 1
    __syncthreads();
    wait_async0();
    compute((T - 1) & 1);              // epilogue only touches acc regs; no barrier needed

    // epilogue: bias (+relu) and store; C frag: VGPR v -> M = lhalf*8+v, N = l16
    #pragma unroll
    for (int mf = 0; mf < 4; ++mf) {
        #pragma unroll
        for (int nf = 0; nf < 2; ++nf) {
            const int gc = n0 + wn * 32 + nf * 16 + l16;
            const float bv = bias[gc];
            #pragma unroll
            for (int v = 0; v < 8; ++v) {
                const int gr = m0 + wm * 64 + mf * 16 + lhalf * 8 + v;
                if (gr < M) {
                    float r = acc[mf][nf][v] + bv;
                    if constexpr (RELU) r = fmaxf(r, 0.f);
                    if constexpr (OUT_BF16) ((__bf16*)Cv)[(size_t)gr * NN + gc] = bfc(r);
                    else                    ((float*)Cv)[(size_t)gr * NN + gc] = r;
                }
            }
        }
    }
}

// ---------------------------------------------------------------------------
// Pooling
// ---------------------------------------------------------------------------
__global__ void pool_init_kernel(float* gsum, unsigned* gmaxb, float* counts) {
    int i = blockIdx.x * 256 + threadIdx.x;
    if (i < NGRAPH * HDIM) { gsum[i] = 0.f; gmaxb[i] = 0xFF800000u; }  // -inf
    if (i < NGRAPH) counts[i] = 0.f;
}

// float atomic max: signed-int max for val>=0, unsigned min for val<0
__device__ inline void atomicMaxF(float* addr, float val) {
    if (val >= 0.f) atomicMax((int*)addr, __float_as_int(val));
    else            atomicMin((unsigned int*)addr, __float_as_uint(val));
}

__global__ void pool_scatter_kernel(const float* __restrict__ h, const int* __restrict__ batch,
                                    float* gsum, unsigned* gmaxb, float* counts) {
    const int n = blockIdx.x, c = threadIdx.x;
    const int b = batch[n];
    const float v = h[(size_t)n * HDIM + c];
    atomicAdd(&gsum[b * HDIM + c], v);
    atomicMaxF((float*)&gmaxb[b * HDIM + c], v);
    if (c == 0) atomicAdd(&counts[b], 1.0f);
}

// ---------------------------------------------------------------------------
// Fused head: gene proj (Linear->LN->ReLU->Linear->ReLU), concat with pooled
// features, 3-layer classifier. One block (256 thr) per graph.
// ---------------------------------------------------------------------------
__global__ __launch_bounds__(256) void head_kernel(
    const float* __restrict__ gene,
    const float* __restrict__ Wg1, const float* __restrict__ bg1,
    const float* __restrict__ lng_g, const float* __restrict__ lng_b,
    const float* __restrict__ Wg2, const float* __restrict__ bg2,
    const float* __restrict__ Wc1, const float* __restrict__ bc1,
    const float* __restrict__ Wc2, const float* __restrict__ bc2,
    const float* __restrict__ Wc3, const float* __restrict__ bc3,
    const float* __restrict__ gsum, const unsigned* __restrict__ gmaxb,
    const float* __restrict__ counts, float* __restrict__ out)
{
    const int gI = blockIdx.x, c = threadIdx.x;
    __shared__ float gvec[640];
    __shared__ float gp1[256];
    __shared__ float h1[256];
    __shared__ float h2[128];
    __shared__ float gf[GFDIM];
    if (c < GFDIM) gf[c] = gene[gI * GFDIM + c];
    __syncthreads();

    float a = bg1[c];
    for (int f = 0; f < GFDIM; ++f) a += gf[f] * Wg1[f * HDIM + c];
    float2 s = block_reduce2(a, a * a);
    float m = s.x * (1.f / HDIM);
    float v = s.y * (1.f / HDIM) - m * m;
    float y = lng_g[c] * (a - m) * rsqrtf(v + 1e-5f) + lng_b[c];
    gp1[c] = fmaxf(y, 0.f);

    const float cnt = counts[gI];
    gvec[c] = gsum[gI * HDIM + c] / fmaxf(cnt, 1.f);
    float mv = __uint_as_float(gmaxb[gI * HDIM + c]);
    if (cnt < 0.5f) mv = 0.f;   // empty segment: -inf -> 0
    gvec[HDIM + c] = mv;
    __syncthreads();

    if (c < 128) {
        float a2 = bg2[c];
        for (int k = 0; k < HDIM; ++k) a2 += gp1[k] * Wg2[k * 128 + c];
        gvec[512 + c] = fmaxf(a2, 0.f);
    }
    __syncthreads();

    {
        float a3 = bc1[c];
        for (int k = 0; k < 640; ++k) a3 += gvec[k] * Wc1[k * HDIM + c];
        h1[c] = fmaxf(a3, 0.f);
    }
    __syncthreads();
    if (c < 128) {
        float a4 = bc2[c];
        for (int k = 0; k < HDIM; ++k) a4 += h1[k] * Wc2[k * 128 + c];
        h2[c] = fmaxf(a4, 0.f);
    }
    __syncthreads();
    if (c < 2) {
        float a5 = bc3[c];
        for (int k = 0; k < 128; ++k) a5 += h2[k] * Wc3[k * 2 + c];
        out[gI * 2 + c] = a5;
    }
}

// ---------------------------------------------------------------------------
extern "C" void kernel_launch(void* const* d_in, const int* in_sizes, int n_in,
                              void* d_out, int out_size, void* d_ws, size_t ws_size,
                              hipStream_t stream)
{
    const float* x     = (const float*)d_in[0];
    const int*   ei    = (const int*)  d_in[1];
    const float* ew    = (const float*)d_in[2];
    const int*   batch = (const int*)  d_in[3];
    const float* gene  = (const float*)d_in[4];
    const float* W_in  = (const float*)d_in[5];
    const float* b_in  = (const float*)d_in[6];
    const float* ln_in_g = (const float*)d_in[7];
    const float* ln_in_b = (const float*)d_in[8];
    const float* eps   = (const float*)d_in[9];
    const float* W1    = (const float*)d_in[10];
    const float* b1    = (const float*)d_in[11];
    const float* W2    = (const float*)d_in[12];
    const float* b2    = (const float*)d_in[13];
    const float* ln_g  = (const float*)d_in[14];
    const float* ln_b  = (const float*)d_in[15];
    const float* Wg1   = (const float*)d_in[16];
    const float* bg1   = (const float*)d_in[17];
    const float* lng_g = (const float*)d_in[18];
    const float* lng_b = (const float*)d_in[19];
    const float* Wg2   = (const float*)d_in[20];
    const float* bg2   = (const float*)d_in[21];
    const float* Wc1   = (const float*)d_in[22];
    const float* bc1   = (const float*)d_in[23];
    const float* Wc2   = (const float*)d_in[24];
    const float* bc2   = (const float*)d_in[25];
    const float* Wc3   = (const float*)d_in[26];
    const float* bc3   = (const float*)d_in[27];
    (void)n_in; (void)out_size; (void)ws_size;

    const int N = in_sizes[0] / FINF;
    const int E = in_sizes[1] / 2;

    char* ws = (char*)d_ws;
    size_t off = 0;
    auto carve = [&](size_t bytes) -> void* {
        void* p = (void*)(ws + off);
        off += (bytes + 255) & ~(size_t)255;
        return p;
    };
    float*    h      = (float*)carve((size_t)N * HDIM * 4);
    float*    agg    = (float*)carve((size_t)N * HDIM * 4);
    void*     Xbf    =          carve((size_t)N * HDIM * 2);    // bf16 agg
    void*     Y      =          carve((size_t)N * HDIM2 * 2);   // bf16 hidden
    float*    Z      = (float*)carve((size_t)N * HDIM * 4);
    void*     W1t    =          carve((size_t)NLAYERS * HDIM * HDIM2 * 2);
    void*     W2t    =          carve((size_t)NLAYERS * HDIM * HDIM2 * 2);
    float*    gsum   = (float*)carve((size_t)NGRAPH * HDIM * 4);
    unsigned* gmaxb  = (unsigned*)carve((size_t)NGRAPH * HDIM * 4);
    float*    counts = (float*)carve((size_t)NGRAPH * 4);

    prep_w_kernel<<<(NLAYERS * HDIM * HDIM2 + 255) / 256, 256, 0, stream>>>(W1, W2, W1t, W2t);
    input_proj_kernel<<<N, 256, 0, stream>>>(x, W_in, b_in, ln_in_g, ln_in_b, h);

    const int gmx = (N + 127) / 128;
    const int cnt4 = N * HDIM / 4;
    for (int l = 0; l < NLAYERS; ++l) {
        agg_init_kernel<<<N, 256, 0, stream>>>(h, eps, l, agg);
        edge_scatter_kernel<<<E, 256, 0, stream>>>(ei, ew, h, agg, E);
        to_bf16_kernel<<<(cnt4 + 255) / 256, 256, 0, stream>>>(agg, Xbf, cnt4);
        gemm_wmma_kernel<HDIM, HDIM2, true, true>
            <<<dim3(gmx, HDIM2 / 128), 256, 0, stream>>>(
                Xbf, (char*)W1t + (size_t)l * HDIM * HDIM2 * 2, b1 + l * HDIM2, Y, N);
        gemm_wmma_kernel<HDIM2, HDIM, false, false>
            <<<dim3(gmx, HDIM / 128), 256, 0, stream>>>(
                Y, (char*)W2t + (size_t)l * HDIM * HDIM2 * 2, b2 + l * HDIM, Z, N);
        ln_res_kernel<<<N, 256, 0, stream>>>(Z, ln_g + l * HDIM, ln_b + l * HDIM, h);
    }

    pool_init_kernel<<<(NGRAPH * HDIM + 255) / 256, 256, 0, stream>>>(gsum, gmaxb, counts);
    pool_scatter_kernel<<<N, 256, 0, stream>>>(h, batch, gsum, gmaxb, counts);
    head_kernel<<<NGRAPH, 256, 0, stream>>>(gene, Wg1, bg1, lng_g, lng_b, Wg2, bg2,
                                            Wc1, bc1, Wc2, bc2, Wc3, bc3,
                                            gsum, gmaxb, counts, (float*)d_out);
}
